// ProposalModule_19705309954645
// MI455X (gfx1250) — matrix operand order repaired and checked by reading, not compiled, not run on hardware
//
#include <hip/hip_runtime.h>
#include <hip/hip_bf16.h>

// ---------------------------------------------------------------------------
// Types for CDNA5 WMMA (wave32): v_wmma_f32_16x16x32_bf16
// ---------------------------------------------------------------------------
typedef __attribute__((ext_vector_type(16))) __bf16 v16bf;
typedef __attribute__((ext_vector_type(8)))  float  v8f;

union BF16x16 { v16bf v; unsigned short u[16]; };

__device__ __forceinline__ unsigned short f2bf(float f) {
    unsigned int x = __float_as_uint(f);
    unsigned int r = x + 0x7FFFu + ((x >> 16) & 1u);   // round-to-nearest-even
    return (unsigned short)(r >> 16);
}

#define NPTS   2048
#define NPOINT 256
#define NSAMP  16
#define BATCH  32
#define NBIG   131072          // B * NPOINT * NSAMP
#define NSMALL 8192            // B * NPOINT
#define RAD2   0.09f
#define INV_RAD (1.0f/0.3f)

// ---------------------------------------------------------------------------
// Fragment layouts used here (contiguous k order, same for A and B), baked into packed global buffers:
//  A (16x32 bf16): lane l -> M = l%16 ; slot s -> K = (l>>4)*16 + s
//  B (32x16 bf16): lane l -> N = l%16 ; slot s -> K = (l>>4)*16 + s
// Packed A: [kt][mtile][lane][slot] ; Packed B: [kt][ntile][lane][slot]
// ---------------------------------------------------------------------------
__device__ __forceinline__ size_t bpack_idx(int c, int n, int ntiles) {
    int kt   = c >> 5;
    int s    = c & 15;
    int lane = ((c >> 4) & 1) * 16 + (n & 15);
    int j    = n >> 4;
    return ((size_t)(kt * ntiles + j) * 32 + lane) * 16 + s;
}

// Weight packer: f32 (M x K) -> bf16 A-fragments.
__global__ void pack_w_kernel(const float* __restrict__ W,
                              unsigned short* __restrict__ Ap,
                              int M, int K, int Kp) {
    int id = blockIdx.x * 256 + threadIdx.x;
    int total = (Kp >> 5) * 4096;
    if (id >= total) return;
    int slot = id & 15;
    int lane = (id >> 4) & 31;
    int mt   = (id >> 9) & 7;
    int kt   = id >> 12;
    int mrow = mt * 16 + (lane & 15);
    int kcol = kt * 32 + (lane >> 4) * 16 + slot;
    float v = (mrow < M && kcol < K) ? W[(size_t)mrow * K + kcol] : 0.0f;
    Ap[id] = f2bf(v);
}

// ---------------------------------------------------------------------------
// Farthest point sampling: one block per batch, 256 sequential iterations.
// ---------------------------------------------------------------------------
__global__ __launch_bounds__(256) void fps_kernel(const float* __restrict__ xyz,
                                                  int* __restrict__ inds,
                                                  float* __restrict__ nxyz) {
    __shared__ float sx[NPTS], sy[NPTS], sz[NPTS], dmin[NPTS];
    __shared__ float rv[256];
    __shared__ int   ri[256];
    __shared__ int   sfar;
    int b = blockIdx.x, t = threadIdx.x;
    for (int k = t; k < NPTS; k += 256) {
        const float* p = xyz + ((size_t)b * NPTS + k) * 3;
        sx[k] = p[0]; sy[k] = p[1]; sz[k] = p[2];
        dmin[k] = 1e10f;
    }
    if (!t) sfar = 0;
    __syncthreads();
    for (int it = 0; it < NPOINT; ++it) {
        int far = sfar;
        if (!t) {
            inds[b * NPOINT + it] = far;
            float* o = nxyz + ((size_t)b * NPOINT + it) * 3;
            o[0] = sx[far]; o[1] = sy[far]; o[2] = sz[far];
        }
        float cx = sx[far], cy = sy[far], cz = sz[far];
        float bv = -1.0f; int bi = 0;
        for (int k = t; k < NPTS; k += 256) {
            float dx = sx[k] - cx, dy = sy[k] - cy, dz = sz[k] - cz;
            float d = dx * dx + dy * dy + dz * dz;
            float nd = fminf(dmin[k], d);
            dmin[k] = nd;
            if (nd > bv) { bv = nd; bi = k; }
        }
        rv[t] = bv; ri[t] = bi;
        __syncthreads();
        for (int off = 128; off; off >>= 1) {
            if (t < off) {
                if (rv[t + off] > rv[t] ||
                    (rv[t + off] == rv[t] && ri[t + off] < ri[t])) {
                    rv[t] = rv[t + off]; ri[t] = ri[t + off];
                }
            }
            __syncthreads();
        }
        if (!t) sfar = ri[0];
        __syncthreads();
    }
}

// ---------------------------------------------------------------------------
// Ball query: ordered first-NSAMP-in-range, padded with first index.
// ---------------------------------------------------------------------------
__global__ __launch_bounds__(256) void ballq_kernel(const float* __restrict__ xyz,
                                                    const float* __restrict__ nxyz,
                                                    int* __restrict__ idx) {
    __shared__ float sx[NPTS], sy[NPTS], sz[NPTS];
    int b = blockIdx.x, t = threadIdx.x;
    for (int k = t; k < NPTS; k += 256) {
        const float* p = xyz + ((size_t)b * NPTS + k) * 3;
        sx[k] = p[0]; sy[k] = p[1]; sz[k] = p[2];
    }
    __syncthreads();
    const float* q = nxyz + ((size_t)b * NPOINT + t) * 3;
    float nx = q[0], ny = q[1], nz = q[2];
    int loc[NSAMP];
    int cnt = 0;
    for (int k = 0; k < NPTS; ++k) {
        float dx = sx[k] - nx, dy = sy[k] - ny, dz = sz[k] - nz;
        if (dx * dx + dy * dy + dz * dz < RAD2) {
            loc[cnt] = k;
            if (++cnt == NSAMP) break;
        }
    }
    int first = (cnt > 0) ? loc[0] : 0;
    for (int s = cnt; s < NSAMP; ++s) loc[s] = first;
    int* o = idx + ((size_t)b * NPOINT + t) * NSAMP;
    for (int s = 0; s < NSAMP; ++s) o[s] = loc[s];
}

// ---------------------------------------------------------------------------
// Generic GEMM (K fixed at 128), LDS-free: Y(128 x N, f32) = A(128x128) * X.
// Block = 8 waves; wave w owns one N-subtile: its B fragment is loaded once
// per k-step (coalesced 32 B/lane); A fragments shared via L2/WGP$.
// All 8 A fragments of a k-step are preloaded into registers BEFORE the
// 8-WMMA burst so loadcnt waits retire incrementally and the fully unrolled
// 4 k-steps can overlap next-step loads with current-step matrix ops.
// ---------------------------------------------------------------------------
__global__ __launch_bounds__(256) void gemm128_packed_kernel(
    const unsigned short* __restrict__ Ap,
    const unsigned short* __restrict__ Xp,
    float* __restrict__ Y, int N) {
    int t = threadIdx.x;
    int w = t >> 5, lane = t & 31;
    int lhalf = lane >> 4, lcol = lane & 15;
    int ntiles = N >> 4;
    int j = blockIdx.x * 8 + w;               // this wave's N-subtile
    v8f zero = {0.f, 0.f, 0.f, 0.f, 0.f, 0.f, 0.f, 0.f};
    v8f acc[8];
#pragma unroll
    for (int mt = 0; mt < 8; ++mt) acc[mt] = zero;
#pragma unroll
    for (int kt = 0; kt < 4; ++kt) {
        BF16x16 bf;
        bf.v = *(const v16bf*)(Xp + ((size_t)(kt * ntiles + j) * 32 + lane) * 16);
        BF16x16 af[8];
#pragma unroll
        for (int mt = 0; mt < 8; ++mt)
            af[mt].v = *(const v16bf*)(Ap + ((size_t)(kt * 8 + mt) * 32 + lane) * 16);
#pragma unroll
        for (int mt = 0; mt < 8; ++mt)
            acc[mt] = __builtin_amdgcn_wmma_f32_16x16x32_bf16(
                false, af[mt].v, false, bf.v, (short)0, acc[mt], false, false);
    }
    int col = j * 16 + lcol;
#pragma unroll
    for (int mt = 0; mt < 8; ++mt) {
#pragma unroll
        for (int r = 0; r < 8; ++r) {
            int m = mt * 16 + r + 8 * lhalf;
            Y[(size_t)m * N + col] = acc[mt][r];
        }
    }
}

// ---------------------------------------------------------------------------
// Layer0 GEMM with fused gather. LDS tile is stored TRANSPOSED (K-contiguous
// per column, stride 40 ushorts to spread banks) so a B fragment is two
// ds_load_b128 per lane. All 8 B fragments are preloaded before the WMMA
// burst so dscnt waits retire incrementally.
// Channels: 0..2 = (grouped_xyz - new_xyz)/R, 3..258 = features, 259..287 = 0.
// ---------------------------------------------------------------------------
#define T_STRIDE 40

__global__ __launch_bounds__(256) void gemm_layer0_kernel(
    const unsigned short* __restrict__ Ap,
    const float* __restrict__ xyz,
    const float* __restrict__ nxyz,
    const int*   __restrict__ idx,
    const float* __restrict__ feat,
    float* __restrict__ Y) {
    __shared__ unsigned short tileT[128 * T_STRIDE];
    int t = threadIdx.x;
    int w = t >> 5, lane = t & 31;
    int lhalf = lane >> 4, lcol = lane & 15;
    int nbase = blockIdx.x * 128;
    int b  = nbase >> 12;          // 4096 positions per batch
    int n0 = nbase & 4095;
    v8f zero = {0.f, 0.f, 0.f, 0.f, 0.f, 0.f, 0.f, 0.f};
    v8f acc[8];
#pragma unroll
    for (int jj = 0; jj < 8; ++jj) acc[jj] = zero;
    for (int kt = 0; kt < 9; ++kt) {
        __syncthreads();
        for (int i = 0; i < 16; ++i) {                // stage 32x128 gathered tile
            int e = t + 256 * i;                      // 4096 elements
            int r = e >> 7, jc = e & 127;             // r = K within tile, jc = col
            int c = kt * 32 + r;
            int n = n0 + jc;
            int p = n >> 4, s = n & 15;
            float v = 0.0f;
            if (c < 259) {
                int k = idx[((size_t)b * NPOINT + p) * NSAMP + s];
                if (c < 3)
                    v = (xyz[((size_t)b * NPTS + k) * 3 + c]
                         - nxyz[((size_t)b * NPOINT + p) * 3 + c]) * INV_RAD;
                else
                    v = feat[((size_t)b * 256 + (c - 3)) * NPTS + k];
            }
            tileT[jc * T_STRIDE + r] = f2bf(v);       // transposed store
        }
        __syncthreads();
        BF16x16 af;
        af.v = *(const v16bf*)(Ap + ((size_t)(kt * 8 + w) * 32 + lane) * 16);
        BF16x16 bf[8];
#pragma unroll
        for (int jj = 0; jj < 8; ++jj) {              // 32 contiguous bytes in LDS
            int col = jj * 16 + lcol;
            bf[jj].v = *(const v16bf*)(tileT + col * T_STRIDE + lhalf * 16);
        }
#pragma unroll
        for (int jj = 0; jj < 8; ++jj)
            acc[jj] = __builtin_amdgcn_wmma_f32_16x16x32_bf16(
                false, af.v, false, bf[jj].v, (short)0, acc[jj], false, false);
    }
#pragma unroll
    for (int jj = 0; jj < 8; ++jj) {
        int col = nbase + jj * 16 + lcol;
#pragma unroll
        for (int r = 0; r < 8; ++r) {
            int m = 16 * w + r + 8 * lhalf;
            Y[(size_t)m * NBIG + col] = acc[jj][r];
        }
    }
}

// ---------------------------------------------------------------------------
// Per-channel batch statistics: mean & 1/sqrt(var+eps) over N positions.
// ---------------------------------------------------------------------------
__global__ __launch_bounds__(256) void stats_kernel(const float* __restrict__ Y,
                                                    float* __restrict__ mean,
                                                    float* __restrict__ invstd,
                                                    int N) {
    __shared__ float s1[256], s2[256];
    int c = blockIdx.x, t = threadIdx.x;
    const float* row = Y + (size_t)c * N;
    float a = 0.f, b = 0.f;
    for (int n = t; n < N; n += 256) { float v = row[n]; a += v; b += v * v; }
    s1[t] = a; s2[t] = b;
    __syncthreads();
    for (int off = 128; off; off >>= 1) {
        if (t < off) { s1[t] += s1[t + off]; s2[t] += s2[t + off]; }
        __syncthreads();
    }
    if (!t) {
        float m = s1[0] / N;
        float var = s2[0] / N - m * m;
        mean[c] = m;
        invstd[c] = rsqrtf(var + 1e-5f);
    }
}

// BN + ReLU -> bf16, written directly in packed B-fragment order.
__global__ __launch_bounds__(256) void bnrelu_kernel(const float* __restrict__ Y,
                                                     unsigned short* __restrict__ Xp,
                                                     const float* __restrict__ mean,
                                                     const float* __restrict__ invstd,
                                                     const float* __restrict__ g,
                                                     const float* __restrict__ be,
                                                     int N) {
    int c = blockIdx.y;
    int n = blockIdx.x * 256 + threadIdx.x;
    float sc = invstd[c] * g[c];
    float sh = be[c] - mean[c] * sc;
    float v = fmaxf(Y[(size_t)c * N + n] * sc + sh, 0.0f);
    Xp[bpack_idx(c, n, N >> 4)] = f2bf(v);
}

// BN + ReLU fused with max-pool over nsample=16 -> packed head input.
__global__ __launch_bounds__(256) void pool_kernel(const float* __restrict__ Y,
                                                   const float* __restrict__ mean,
                                                   const float* __restrict__ invstd,
                                                   const float* __restrict__ g,
                                                   const float* __restrict__ be,
                                                   unsigned short* __restrict__ Xp) {
    int c = blockIdx.y;
    int q = blockIdx.x * 256 + threadIdx.x;   // b*256+p
    float sc = invstd[c] * g[c];
    float sh = be[c] - mean[c] * sc;
    const float* src = Y + (size_t)c * NBIG + (size_t)q * NSAMP;
    float mx = -3.4e38f;
    for (int s = 0; s < NSAMP; ++s) mx = fmaxf(mx, src[s] * sc + sh);
    mx = fmaxf(mx, 0.0f);                     // relu(max) == max(relu)
    Xp[bpack_idx(c, q, NSMALL >> 4)] = f2bf(mx);
}

// Final assembly: out(B,P,122) = [net_t(119) | new_xyz + net_t[2:5]]
__global__ void assemble_kernel(const float* __restrict__ Y3,
                                const float* __restrict__ b3,
                                const float* __restrict__ nxyz,
                                float* __restrict__ out) {
    int q = blockIdx.x;          // 0..8191
    int j = threadIdx.x;         // 128 threads, 122 active
    if (j >= 122) return;
    float v;
    if (j < 119) {
        v = Y3[(size_t)j * NSMALL + q] + b3[j];
    } else {
        int d = j - 119;
        v = nxyz[(size_t)q * 3 + d] + Y3[(size_t)(2 + d) * NSMALL + q] + b3[2 + d];
    }
    out[(size_t)q * 122 + j] = v;
}

// ---------------------------------------------------------------------------
// Host orchestration
// ---------------------------------------------------------------------------
extern "C" void kernel_launch(void* const* d_in, const int* in_sizes, int n_in,
                              void* d_out, int out_size, void* d_ws, size_t ws_size,
                              hipStream_t stream) {
    const float* xyz   = (const float*)d_in[0];
    const float* feat  = (const float*)d_in[1];
    const float* w0    = (const float*)d_in[2];
    const float* g0    = (const float*)d_in[3];
    const float* be0   = (const float*)d_in[4];
    const float* w1m   = (const float*)d_in[5];
    const float* g1m   = (const float*)d_in[6];
    const float* be1m  = (const float*)d_in[7];
    const float* w2m   = (const float*)d_in[8];
    const float* g2m   = (const float*)d_in[9];
    const float* be2m  = (const float*)d_in[10];
    const float* hw1   = (const float*)d_in[11];
    // d_in[12] = b1: cancels under batch-norm (x+b - mean(x+b) == x - mean(x))
    const float* hg1   = (const float*)d_in[13];
    const float* hbe1  = (const float*)d_in[14];
    const float* hw2   = (const float*)d_in[15];
    // d_in[16] = b2: cancels under batch-norm
    const float* hg2   = (const float*)d_in[17];
    const float* hbe2  = (const float*)d_in[18];
    const float* w3    = (const float*)d_in[19];
    const float* b3    = (const float*)d_in[20];
    float* out = (float*)d_out;

    char* ws = (char*)d_ws;
    size_t off = 0;
    auto take = [&](size_t bytes) -> void* {
        void* p = ws + off;
        off = (off + bytes + 255) & ~(size_t)255;
        return p;
    };
    int*   fps_inds = (int*)  take((size_t)BATCH * NPOINT * 4);
    float* nxyz     = (float*)take((size_t)BATCH * NPOINT * 3 * 4);
    int*   idx      = (int*)  take((size_t)BATCH * NPOINT * NSAMP * 4);
    unsigned short* ApL0 = (unsigned short*)take((size_t)9 * 4096 * 2);
    unsigned short* Ap1  = (unsigned short*)take((size_t)4 * 4096 * 2);
    unsigned short* Ap2  = (unsigned short*)take((size_t)4 * 4096 * 2);
    unsigned short* ApH1 = (unsigned short*)take((size_t)4 * 4096 * 2);
    unsigned short* ApH2 = (unsigned short*)take((size_t)4 * 4096 * 2);
    unsigned short* ApW3 = (unsigned short*)take((size_t)4 * 4096 * 2);
    float* mean   = (float*)take(128 * 4);
    float* invstd = (float*)take(128 * 4);
    float* Ybuf   = (float*)take((size_t)128 * NBIG * 4);          // 64 MB
    unsigned short* Xbuf = (unsigned short*)take((size_t)128 * NBIG * 2); // 33.5 MB
    unsigned short* Xh  = Xbuf;                 // reuse: head buffers (2 MB each)
    unsigned short* Xh2 = Xbuf + (1u << 20);
    unsigned short* Xh3 = Xbuf + (2u << 20);

    // Pack weights into bf16 A-fragment order.
    pack_w_kernel<<<(9 * 4096 + 255) / 256, 256, 0, stream>>>(w0,  ApL0, 128, 259, 288);
    pack_w_kernel<<<(4 * 4096 + 255) / 256, 256, 0, stream>>>(w1m, Ap1, 128, 128, 128);
    pack_w_kernel<<<(4 * 4096 + 255) / 256, 256, 0, stream>>>(w2m, Ap2, 128, 128, 128);
    pack_w_kernel<<<(4 * 4096 + 255) / 256, 256, 0, stream>>>(hw1, ApH1, 128, 128, 128);
    pack_w_kernel<<<(4 * 4096 + 255) / 256, 256, 0, stream>>>(hw2, ApH2, 128, 128, 128);
    pack_w_kernel<<<(4 * 4096 + 255) / 256, 256, 0, stream>>>(w3,  ApW3, 119, 128, 128);

    // Sampling + grouping.
    fps_kernel  <<<BATCH, 256, 0, stream>>>(xyz, fps_inds, nxyz);
    ballq_kernel<<<BATCH, 256, 0, stream>>>(xyz, nxyz, idx);

    // MLP layer 0 (fused gather) -> BN -> ReLU (packed)
    gemm_layer0_kernel<<<NBIG / 128, 256, 0, stream>>>(ApL0, xyz, nxyz, idx, feat, Ybuf);
    stats_kernel<<<128, 256, 0, stream>>>(Ybuf, mean, invstd, NBIG);
    bnrelu_kernel<<<dim3(NBIG / 256, 128), 256, 0, stream>>>(Ybuf, Xbuf, mean, invstd, g0, be0, NBIG);

    // MLP layer 1
    gemm128_packed_kernel<<<NBIG / 128, 256, 0, stream>>>(Ap1, Xbuf, Ybuf, NBIG);
    stats_kernel<<<128, 256, 0, stream>>>(Ybuf, mean, invstd, NBIG);
    bnrelu_kernel<<<dim3(NBIG / 256, 128), 256, 0, stream>>>(Ybuf, Xbuf, mean, invstd, g1m, be1m, NBIG);

    // MLP layer 2 -> BN -> ReLU fused with max-pool over nsample
    gemm128_packed_kernel<<<NBIG / 128, 256, 0, stream>>>(Ap2, Xbuf, Ybuf, NBIG);
    stats_kernel<<<128, 256, 0, stream>>>(Ybuf, mean, invstd, NBIG);
    pool_kernel<<<dim3(NSMALL / 256, 128), 256, 0, stream>>>(Ybuf, mean, invstd, g2m, be2m, Xh);

    // Head layer 1
    gemm128_packed_kernel<<<NSMALL / 128, 256, 0, stream>>>(ApH1, Xh, Ybuf, NSMALL);
    stats_kernel<<<128, 256, 0, stream>>>(Ybuf, mean, invstd, NSMALL);
    bnrelu_kernel<<<dim3(NSMALL / 256, 128), 256, 0, stream>>>(Ybuf, Xh2, mean, invstd, hg1, hbe1, NSMALL);

    // Head layer 2
    gemm128_packed_kernel<<<NSMALL / 128, 256, 0, stream>>>(ApH2, Xh2, Ybuf, NSMALL);
    stats_kernel<<<128, 256, 0, stream>>>(Ybuf, mean, invstd, NSMALL);
    bnrelu_kernel<<<dim3(NSMALL / 256, 128), 256, 0, stream>>>(Ybuf, Xh3, mean, invstd, hg2, hbe2, NSMALL);

    // Final projection + decode/concat
    gemm128_packed_kernel<<<NSMALL / 128, 256, 0, stream>>>(ApW3, Xh3, Ybuf, NSMALL);
    assemble_kernel<<<NSMALL, 128, 0, stream>>>(Ybuf, b3, nxyz, out);
}
